// GraphTransformerBlock_69131793596604
// MI455X (gfx1250) — compile-verified, matrix operand order checked by reading, not run on hardware
//
#include <hip/hip_runtime.h>
#include <hip/hip_bf16.h>

typedef __bf16 bf16;
typedef __attribute__((ext_vector_type(16))) __bf16 v16bf;
typedef __attribute__((ext_vector_type(8)))  __bf16 v8bf;
typedef __attribute__((ext_vector_type(8)))  float  v8f;
typedef __attribute__((ext_vector_type(4)))  int    v4i;

#define NB 2
#define NN 256
#define NODE_H 256
#define EDGE_H 128
#define HD 256   // H*D
#define NH 8
#define DD 32

#if __has_builtin(__builtin_amdgcn_global_load_async_to_lds_b128) && \
    __has_builtin(__builtin_amdgcn_s_wait_asynccnt)
#define ASYNC_OK 1
#else
#define ASYNC_OK 0
#endif

typedef __attribute__((address_space(1))) v4i* gv4i_p;
typedef __attribute__((address_space(3))) v4i* lv4i_p;

__device__ __forceinline__ bf16 f2bf(float f) { return (bf16)f; }
__device__ __forceinline__ float bf2f(bf16 h) { return (float)h; }
__device__ __forceinline__ float lrelu(float x) { return x > 0.f ? x : 0.01f * x; }

__device__ __forceinline__ v8f wmma_bf16(v16bf a, v16bf b, v8f c) {
  return __builtin_amdgcn_wmma_f32_16x16x32_bf16(false, a, false, b, (short)0, c, false, false);
}

// K index for element-pair v (vgpr 0..7) on lane-half h, per ISA 7.12.2 16-bit A layout
__device__ __forceinline__ int kmap(int v, int h) {
  return (v < 4) ? (2 * v + 8 * h) : (16 + 2 * (v - 4) + 8 * h);
}

// A fragment (16xK tile, row-major bf16 in LDS, leading dim ldA), k0 = K offset.
// Per ISA layout this is two contiguous 16-byte runs per lane -> 2x ds_load_b128.
__device__ __forceinline__ v16bf load_a_lds(const bf16* A, int ldA, int k0, int lane) {
  const bf16* row = A + (lane & 15) * ldA + k0 + 8 * (lane >> 4);
  v8bf lo = *(const v8bf*)row;         // K = 8h .. 8h+7
  v8bf hi = *(const v8bf*)(row + 16);  // K = 16+8h .. 16+8h+7
  return __builtin_shufflevector(lo, hi, 0, 1, 2, 3, 4, 5, 6, 7, 8, 9, 10, 11, 12, 13, 14, 15);
}

// B fragment from fragment-swizzled weight: tile (k0/32, n0/16), 32 bytes per lane,
// contiguous -> 2x global_load_b128.
__device__ __forceinline__ v16bf load_b_frag(const bf16* Wf, int ntiles, int k0, int n0, int lane) {
  const bf16* p = Wf + ((((k0 >> 5) * ntiles + (n0 >> 4)) << 5) + lane) * 16;
  return *(const v16bf*)p;
}

// store C/D accumulator tile to LDS as bf16 (row-major, leading dim ldo)
__device__ __forceinline__ void store_acc_lds_bf16(bf16* out, int ldo, int n0, int lane, const v8f& acc) {
  int half = lane >> 4, col = n0 + (lane & 15);
#pragma unroll
  for (int r = 0; r < 8; ++r) out[(r + 8 * half) * ldo + col] = f2bf(acc[r]);
}

// LayerNorm over 16 rows of C fp32 values in LDS -> bf16 out. 256 threads.
__device__ __forceinline__ void ln16(const float* buf, int ldb, int C,
                                     const float* g, const float* bb,
                                     bf16* out, int ldo, int tid,
                                     float (*red1)[16], float (*red2)[16]) {
  int r = tid >> 4, c0 = tid & 15;
  float s = 0.f, s2 = 0.f;
  for (int c = c0; c < C; c += 16) { float x = buf[r * ldb + c]; s += x; s2 += x * x; }
  red1[r][c0] = s; red2[r][c0] = s2;
  __syncthreads();
  if (c0 == 0) {
    float ts = 0.f, ts2 = 0.f;
    for (int t = 0; t < 16; ++t) { ts += red1[r][t]; ts2 += red2[r][t]; }
    float mean = ts / C;
    float var = ts2 / C - mean * mean;
    red1[r][0] = mean;
    red2[r][0] = rsqrtf(var + 1e-5f);
  }
  __syncthreads();
  float mean = red1[r][0], rstd = red2[r][0];
  for (int c = c0; c < C; c += 16)
    out[r * ldo + c] = f2bf((buf[r * ldb + c] - mean) * rstd * g[c] + bb[c]);
  __syncthreads();
}

// ---------------- K0: weight prep (fp32 -> fragment-swizzled bf16, Wsum) ----------------
// Output layout: frag[kt*ntiles + nt][lane(32)][elem(16)] where lane's 16 bf16 values are
// exactly the v16bf B operand for WMMA tile (kt, nt).
__device__ __forceinline__ void swizzleB(const float* W, bf16* out, int K, int Ncols,
                                         int t, int stride) {
  int ntiles = Ncols >> 4;
  int total = K * Ncols;
  for (int idx = t; idx < total; idx += stride) {
    int j = idx & 15;
    int lane = (idx >> 4) & 31;
    int tile = idx >> 9;
    int nt = tile % ntiles;
    int kt = tile / ntiles;
    int h = lane >> 4;
    int v = j >> 1;
    int k = (kt << 5) + kmap(v, h) + (j & 1);
    int n = (nt << 4) + (lane & 15);
    out[idx] = f2bf(W[k * Ncols + n]);
  }
}

__global__ void prep_kernel(const float* Wqk, const float* Wedge, const float* Wv,
                            const float* Wno, const float* Weo, const float* W1x, const float* W2x,
                            const float* W1e, const float* W2e,
                            bf16* bWqk, bf16* bWedge, bf16* bWv, bf16* bWno, bf16* bWeo,
                            bf16* bW1x, bf16* bW2x, bf16* bW1e, bf16* bW2e, float* Wsum) {
  int t = blockIdx.x * blockDim.x + threadIdx.x;
  int stride = gridDim.x * blockDim.x;
  swizzleB(Wqk, bWqk, NODE_H, 2 * HD, t, stride);
  swizzleB(Wedge, bWedge, EDGE_H, HD, t, stride);
  swizzleB(Wv, bWv, NODE_H, HD, t, stride);
  swizzleB(Wno, bWno, HD, NODE_H, t, stride);
  swizzleB(Weo, bWeo, HD, EDGE_H, t, stride);
  swizzleB(W1x, bW1x, NODE_H, NODE_H, t, stride);
  swizzleB(W2x, bW2x, NODE_H, NODE_H, t, stride);
  swizzleB(W1e, bW1e, EDGE_H, EDGE_H, t, stride);
  swizzleB(W2e, bW2e, EDGE_H, EDGE_H, t, stride);
  for (int i = t; i < EDGE_H * NH; i += stride) {
    int f = i >> 3, h = i & 7;
    float s = 0.f;
    for (int d = 0; d < DD; ++d) s += Wedge[f * HD + h * DD + d];
    Wsum[i] = s;
  }
}

// ---------------- K1: node LN + qk/v projections (WMMA) ----------------
__global__ __launch_bounds__(256) void node_qkv_kernel(
    const float* __restrict__ x, const float* __restrict__ g_ln_n, const float* __restrict__ b_ln_n,
    const float* __restrict__ bv, const bf16* __restrict__ bWqk, const bf16* __restrict__ bWv,
    float* __restrict__ qkout, float* __restrict__ vout) {
  __shared__ __align__(32) float xb[16][NODE_H];
  __shared__ __align__(32) bf16 xnb[16][NODE_H];
  __shared__ float red1[16][16], red2[16][16];
  int tid = threadIdx.x, lane = tid & 31, wave = tid >> 5;
  long row0 = (long)blockIdx.x * 16;
  for (int idx = tid; idx < (16 * NODE_H) / 4; idx += 256)
    ((float4*)&xb[0][0])[idx] = ((const float4*)(x + row0 * NODE_H))[idx];
  __syncthreads();
  ln16(&xb[0][0], NODE_H, NODE_H, g_ln_n, b_ln_n, &xnb[0][0], NODE_H, tid, red1, red2);
  // 48 output tiles of 16 cols: [0,512) -> qk, [512,768) -> v
  for (int t = 0; t < 6; ++t) {
    int n0 = (wave * 6 + t) * 16;
    v8f acc = {};
    if (n0 < 2 * HD) {
      for (int k0 = 0; k0 < NODE_H; k0 += 32) {
        v16bf a = load_a_lds(&xnb[0][0], NODE_H, k0, lane);
        v16bf bm = load_b_frag(bWqk, 32, k0, n0, lane);
        acc = wmma_bf16(a, bm, acc);
      }
      int half = lane >> 4, col = n0 + (lane & 15);
#pragma unroll
      for (int r = 0; r < 8; ++r)
        qkout[(row0 + r + 8 * half) * (2 * HD) + col] = acc[r];
    } else {
      int c0 = n0 - 2 * HD;
      for (int k0 = 0; k0 < NODE_H; k0 += 32) {
        v16bf a = load_a_lds(&xnb[0][0], NODE_H, k0, lane);
        v16bf bm = load_b_frag(bWv, 16, k0, c0, lane);
        acc = wmma_bf16(a, bm, acc);
      }
      int half = lane >> 4, col = c0 + (lane & 15);
      float bvv = bv[col];
#pragma unroll
      for (int r = 0; r < 8; ++r)
        vout[(row0 + r + 8 * half) * HD + col] = acc[r] + bvv;
    }
  }
}

// ---------------- K2: fused edge path (LN, e-GEMM, +q*k, Weo-GEMM, edge FFN) ----------------
__global__ __launch_bounds__(256) void edge_kernel(
    const float* __restrict__ edge_attr, const float* __restrict__ qk,
    const float* __restrict__ g_ln_e, const float* __restrict__ b_ln_e,
    const float* __restrict__ beo, const float* __restrict__ g_fe, const float* __restrict__ b_fe,
    const float* __restrict__ b1e, const float* __restrict__ b2e,
    const bf16* __restrict__ bWedge, const bf16* __restrict__ bWeo,
    const bf16* __restrict__ bW1e, const bf16* __restrict__ bW2e,
    const float* __restrict__ Wsum,
    float* __restrict__ att_part, float* __restrict__ out_edge) {
  __shared__ __align__(32) float es[16][EDGE_H];    // original edge tile (residual)
  __shared__ __align__(32) bf16 enb[16][EDGE_H];    // LN'd edge tile; reused for FFN LN output
  __shared__ __align__(32) bf16 rawb[16][HD];       // raw_att tile in bf16
  __shared__ __align__(32) float eaf[16][EDGE_H];   // ea = edge_out + residual
  __shared__ __align__(32) bf16 h1b[16][EDGE_H];    // FFN hidden
  __shared__ float red1[16][16], red2[16][16];

  const int tid = threadIdx.x, lane = tid & 31, wave = tid >> 5;
  const int wg = blockIdx.x;
  const int j0 = (wg & 15) * 16;
  const int i = (wg >> 4) & 255;
  const int b = wg >> 12;
  const long base = ((long)(b * NN + i) * NN + j0) * EDGE_H;

#if ASYNC_OK
  {
    char* gsrc = (char*)(edge_attr + base);
    char* ldst = (char*)&es[0][0];
    for (int idx = tid; idx < (16 * EDGE_H) / 4; idx += 256) {
      __builtin_amdgcn_global_load_async_to_lds_b128(
          (gv4i_p)(gsrc + idx * 16), (lv4i_p)(ldst + idx * 16), 0, 0);
    }
    __builtin_amdgcn_s_wait_asynccnt(0);
  }
#else
  for (int idx = tid; idx < (16 * EDGE_H) / 4; idx += 256)
    ((float4*)&es[0][0])[idx] = ((const float4*)(edge_attr + base))[idx];
#endif
  __syncthreads();

  ln16(&es[0][0], EDGE_H, EDGE_H, g_ln_e, b_ln_e, &enb[0][0], EDGE_H, tid, red1, red2);

  // attention logit partial: en . Wsum (per row r, head h)
  if (tid < 128) {
    int r = tid >> 3, h = tid & 7;
    float s = 0.f;
    for (int f = 0; f < EDGE_H; ++f) s += bf2f(enb[r][f]) * Wsum[f * NH + h];
    att_part[(((long)(b * NN + i) * NN) + j0 + r) * NH + h] = s;
  }

  // GEMM1: e(16x256) = en(16x128) @ Wedge(128x256); two 16-col tiles per wave; then add q*k
  {
    int n0a = (wave * 2) * 16, n0b = (wave * 2 + 1) * 16;
    v8f acc0 = {}, acc1 = {};
    for (int k0 = 0; k0 < EDGE_H; k0 += 32) {
      v16bf a = load_a_lds(&enb[0][0], EDGE_H, k0, lane);
      v16bf b0 = load_b_frag(bWedge, 16, k0, n0a, lane);
      v16bf b1 = load_b_frag(bWedge, 16, k0, n0b, lane);
      acc0 = wmma_bf16(a, b0, acc0);
      acc1 = wmma_bf16(a, b1, acc1);
    }
    int half = lane >> 4;
    {
      int c = n0a + (lane & 15);
      float qv = qk[((long)(b * NN + i)) * (2 * HD) + c];
#pragma unroll
      for (int r = 0; r < 8; ++r) {
        int j = j0 + r + 8 * half;
        acc0[r] += qv * qk[((long)(b * NN + j)) * (2 * HD) + HD + c];
      }
    }
    {
      int c = n0b + (lane & 15);
      float qv = qk[((long)(b * NN + i)) * (2 * HD) + c];
#pragma unroll
      for (int r = 0; r < 8; ++r) {
        int j = j0 + r + 8 * half;
        acc1[r] += qv * qk[((long)(b * NN + j)) * (2 * HD) + HD + c];
      }
    }
    store_acc_lds_bf16(&rawb[0][0], HD, n0a, lane, acc0);
    store_acc_lds_bf16(&rawb[0][0], HD, n0b, lane, acc1);
  }
  __syncthreads();

  // GEMM2: edge_out(16x128) = raw(16x256) @ Weo(256x128); one 16-col tile per wave
  const int n0 = wave * 16;
  {
    v8f acc = {};
    for (int k0 = 0; k0 < HD; k0 += 32) {
      v16bf a = load_a_lds(&rawb[0][0], HD, k0, lane);
      v16bf bm = load_b_frag(bWeo, 8, k0, n0, lane);
      acc = wmma_bf16(a, bm, acc);
    }
    int half = lane >> 4, col = n0 + (lane & 15);
    float bc = beo[col];
#pragma unroll
    for (int r = 0; r < 8; ++r) {
      int row = r + 8 * half;
      eaf[row][col] = acc[r] + bc + es[row][col];
    }
  }
  __syncthreads();

  // edge FFN: LN -> W1e (lrelu) -> W2e (lrelu) -> +ea
  ln16(&eaf[0][0], EDGE_H, EDGE_H, g_fe, b_fe, &enb[0][0], EDGE_H, tid, red1, red2);
  {
    v8f acc = {};
    for (int k0 = 0; k0 < EDGE_H; k0 += 32) {
      v16bf a = load_a_lds(&enb[0][0], EDGE_H, k0, lane);
      v16bf bm = load_b_frag(bW1e, 8, k0, n0, lane);
      acc = wmma_bf16(a, bm, acc);
    }
    int half = lane >> 4, col = n0 + (lane & 15);
    float bc = b1e[col];
#pragma unroll
    for (int r = 0; r < 8; ++r)
      h1b[r + 8 * half][col] = f2bf(lrelu(acc[r] + bc));
  }
  __syncthreads();
  {
    v8f acc = {};
    for (int k0 = 0; k0 < EDGE_H; k0 += 32) {
      v16bf a = load_a_lds(&h1b[0][0], EDGE_H, k0, lane);
      v16bf bm = load_b_frag(bW2e, 8, k0, n0, lane);
      acc = wmma_bf16(a, bm, acc);
    }
    int half = lane >> 4, col = n0 + (lane & 15);
    float bc = b2e[col];
#pragma unroll
    for (int r = 0; r < 8; ++r) {
      int row = r + 8 * half;
      out_edge[base + row * EDGE_H + col] = lrelu(acc[r] + bc) + eaf[row][col];
    }
  }
}

// ---------------- K3: softmax attention + A@V ----------------
__global__ __launch_bounds__(256) void attn_kernel(
    const float* __restrict__ att_part, const float* __restrict__ qk,
    const float* __restrict__ vb, const float* __restrict__ mask,
    float* __restrict__ node_att) {
  __shared__ float logit[NN][NH];
  __shared__ float qrow[NN];
  __shared__ float redm[32][NH], reds[32][NH];
  __shared__ float mxs[NH], sms[NH];
  int tid = threadIdx.x;
  int b = blockIdx.x >> 8, i = blockIdx.x & 255;
  qrow[tid] = qk[((long)(b * NN + i)) * (2 * HD) + tid];
  __syncthreads();
  int j = tid;
  const float scale = 0.17677669529663687f;  // 1/sqrt(32)
  const float* krow = qk + ((long)(b * NN + j)) * (2 * HD) + HD;
  const float* ap = att_part + (((long)(b * NN + i) * NN) + j) * NH;
  bool valid = mask[b * NN + j] != 0.0f;
#pragma unroll
  for (int h = 0; h < NH; ++h) {
    float dot = 0.f;
#pragma unroll
    for (int d = 0; d < DD; ++d) dot += qrow[h * DD + d] * krow[h * DD + d];
    float l = (dot + ap[h]) * scale;
    logit[j][h] = valid ? l : -3.0e38f;
  }
  __syncthreads();
  // parallel max: 32 groups of 8 rows per head
  {
    int h = tid & 7, g = tid >> 3;
    float mx = -3.0e38f;
    for (int jj = g * 8; jj < g * 8 + 8; ++jj) mx = fmaxf(mx, logit[jj][h]);
    redm[g][h] = mx;
  }
  __syncthreads();
  if (tid < NH) {
    float m = -3.0e38f;
    for (int g = 0; g < 32; ++g) m = fmaxf(m, redm[g][tid]);
    mxs[tid] = m;
  }
  __syncthreads();
  {
    int h = tid & 7, g = tid >> 3;
    float sm = 0.f;
    for (int jj = g * 8; jj < g * 8 + 8; ++jj) sm += __expf(logit[jj][h] - mxs[h]);
    reds[g][h] = sm;
  }
  __syncthreads();
  if (tid < NH) {
    float s = 0.f;
    for (int g = 0; g < 32; ++g) s += reds[g][tid];
    sms[tid] = 1.0f / s;
  }
  __syncthreads();
#pragma unroll
  for (int h = 0; h < NH; ++h)
    logit[j][h] = __expf(logit[j][h] - mxs[h]) * sms[h];
  __syncthreads();
  int c = tid, h = c >> 5;
  float acc = 0.f;
  for (int jj = 0; jj < NN; ++jj)
    acc += logit[jj][h] * vb[((long)(b * NN + jj)) * HD + c];
  node_att[((long)(b * NN + i)) * HD + c] = acc;
}

// ---------------- K4: node out-proj + residual + node FFN (WMMA) ----------------
__global__ __launch_bounds__(256) void node_out_kernel(
    const float* __restrict__ node_att, const float* __restrict__ x,
    const float* __restrict__ bno, const float* __restrict__ g_fx, const float* __restrict__ b_fx,
    const float* __restrict__ b1x, const float* __restrict__ b2x,
    const bf16* __restrict__ bWno, const bf16* __restrict__ bW1x, const bf16* __restrict__ bW2x,
    float* __restrict__ out_x) {
  __shared__ __align__(32) bf16 ab[16][NODE_H];
  __shared__ __align__(32) bf16 hb[16][NODE_H];
  __shared__ __align__(32) float x1[16][NODE_H];
  __shared__ float red1[16][16], red2[16][16];
  int tid = threadIdx.x, lane = tid & 31, wave = tid >> 5;
  long row0 = (long)blockIdx.x * 16;
  for (int idx = tid; idx < 16 * NODE_H; idx += 256)
    (&ab[0][0])[idx] = f2bf(node_att[row0 * NODE_H + idx]);
  __syncthreads();
  // x1 = node_att @ Wno + bno + x
  for (int t = 0; t < 2; ++t) {
    int n0 = (wave * 2 + t) * 16;
    v8f acc = {};
    for (int k0 = 0; k0 < HD; k0 += 32) {
      v16bf a = load_a_lds(&ab[0][0], NODE_H, k0, lane);
      v16bf bm = load_b_frag(bWno, 16, k0, n0, lane);
      acc = wmma_bf16(a, bm, acc);
    }
    int half = lane >> 4, col = n0 + (lane & 15);
    float bc = bno[col];
#pragma unroll
    for (int r = 0; r < 8; ++r) {
      int row = r + 8 * half;
      x1[row][col] = acc[r] + bc + x[(row0 + row) * NODE_H + col];
    }
  }
  __syncthreads();
  ln16(&x1[0][0], NODE_H, NODE_H, g_fx, b_fx, &hb[0][0], NODE_H, tid, red1, red2);
  // h = lrelu(hn @ W1x + b1x)  -> ab
  for (int t = 0; t < 2; ++t) {
    int n0 = (wave * 2 + t) * 16;
    v8f acc = {};
    for (int k0 = 0; k0 < NODE_H; k0 += 32) {
      v16bf a = load_a_lds(&hb[0][0], NODE_H, k0, lane);
      v16bf bm = load_b_frag(bW1x, 16, k0, n0, lane);
      acc = wmma_bf16(a, bm, acc);
    }
    int half = lane >> 4, col = n0 + (lane & 15);
    float bc = b1x[col];
#pragma unroll
    for (int r = 0; r < 8; ++r)
      ab[r + 8 * half][col] = f2bf(lrelu(acc[r] + bc));
  }
  __syncthreads();
  // out = lrelu(h @ W2x + b2x) + x1
  for (int t = 0; t < 2; ++t) {
    int n0 = (wave * 2 + t) * 16;
    v8f acc = {};
    for (int k0 = 0; k0 < NODE_H; k0 += 32) {
      v16bf a = load_a_lds(&ab[0][0], NODE_H, k0, lane);
      v16bf bm = load_b_frag(bW2x, 16, k0, n0, lane);
      acc = wmma_bf16(a, bm, acc);
    }
    int half = lane >> 4, col = n0 + (lane & 15);
    float bc = b2x[col];
#pragma unroll
    for (int r = 0; r < 8; ++r) {
      int row = r + 8 * half;
      out_x[(row0 + row) * NODE_H + col] = lrelu(acc[r] + bc) + x1[row][col];
    }
  }
}

extern "C" void kernel_launch(void* const* d_in, const int* in_sizes, int n_in,
                              void* d_out, int out_size, void* d_ws, size_t ws_size,
                              hipStream_t stream) {
  (void)in_sizes; (void)n_in; (void)out_size; (void)ws_size;
  const float* x = (const float*)d_in[0];
  const float* edge_attr = (const float*)d_in[1];
  const float* mask = (const float*)d_in[2];
  const float* Wqk = (const float*)d_in[3];
  const float* Wedge = (const float*)d_in[4];
  const float* Wv = (const float*)d_in[5];
  const float* bv = (const float*)d_in[6];
  const float* Wno = (const float*)d_in[7];
  const float* bno = (const float*)d_in[8];
  const float* Weo = (const float*)d_in[9];
  const float* beo = (const float*)d_in[10];
  const float* g_ln_n = (const float*)d_in[11];
  const float* b_ln_n = (const float*)d_in[12];
  const float* g_ln_e = (const float*)d_in[13];
  const float* b_ln_e = (const float*)d_in[14];
  const float* g_fx = (const float*)d_in[15];
  const float* b_fx = (const float*)d_in[16];
  const float* W1x = (const float*)d_in[17];
  const float* b1x = (const float*)d_in[18];
  const float* W2x = (const float*)d_in[19];
  const float* b2x = (const float*)d_in[20];
  const float* g_fe = (const float*)d_in[21];
  const float* b_fe = (const float*)d_in[22];
  const float* W1e = (const float*)d_in[23];
  const float* b1e = (const float*)d_in[24];
  const float* W2e = (const float*)d_in[25];
  const float* b2e = (const float*)d_in[26];

  float* out_x = (float*)d_out;
  float* out_e = out_x + (size_t)NB * NN * NODE_H;

  char* w = (char*)d_ws;
  auto alloc = [&](size_t bytes) {
    void* p = (void*)w;
    w += (bytes + 255) & ~(size_t)255;
    return p;
  };
  float* qk       = (float*)alloc((size_t)NB * NN * 2 * HD * 4);
  float* vbuf     = (float*)alloc((size_t)NB * NN * HD * 4);
  float* att_part = (float*)alloc((size_t)NB * NN * NN * NH * 4);
  float* node_att = (float*)alloc((size_t)NB * NN * HD * 4);
  float* Wsum     = (float*)alloc((size_t)EDGE_H * NH * 4);
  bf16* bWqk   = (bf16*)alloc((size_t)NODE_H * 2 * HD * 2);
  bf16* bWedge = (bf16*)alloc((size_t)EDGE_H * HD * 2);
  bf16* bWv    = (bf16*)alloc((size_t)NODE_H * HD * 2);
  bf16* bWno   = (bf16*)alloc((size_t)HD * NODE_H * 2);
  bf16* bWeo   = (bf16*)alloc((size_t)HD * EDGE_H * 2);
  bf16* bW1x   = (bf16*)alloc((size_t)NODE_H * NODE_H * 2);
  bf16* bW2x   = (bf16*)alloc((size_t)NODE_H * NODE_H * 2);
  bf16* bW1e   = (bf16*)alloc((size_t)EDGE_H * EDGE_H * 2);
  bf16* bW2e   = (bf16*)alloc((size_t)EDGE_H * EDGE_H * 2);

  prep_kernel<<<256, 256, 0, stream>>>(Wqk, Wedge, Wv, Wno, Weo, W1x, W2x, W1e, W2e,
                                       bWqk, bWedge, bWv, bWno, bWeo, bW1x, bW2x, bW1e, bW2e, Wsum);
  node_qkv_kernel<<<(NB * NN) / 16, 256, 0, stream>>>(x, g_ln_n, b_ln_n, bv, bWqk, bWv, qk, vbuf);
  edge_kernel<<<(NB * NN * NN) / 16, 256, 0, stream>>>(edge_attr, qk, g_ln_e, b_ln_e, beo,
                                                       g_fe, b_fe, b1e, b2e,
                                                       bWedge, bWeo, bW1e, bW2e, Wsum,
                                                       att_part, out_e);
  attn_kernel<<<NB * NN, 256, 0, stream>>>(att_part, qk, vbuf, mask, node_att);
  node_out_kernel<<<(NB * NN) / 16, 256, 0, stream>>>(node_att, x, bno, g_fx, b_fx, b1x, b2x,
                                                      bWno, bW1x, bW2x, out_x);
}
